// CABlock_43971875176865
// MI455X (gfx1250) — matrix operand order repaired and checked by reading, not compiled, not run on hardware
//
#include <hip/hip_runtime.h>
#include <hip/hip_bf16.h>

typedef __bf16 bf16_t;
typedef __attribute__((ext_vector_type(16))) __bf16 v16bf;
typedef __attribute__((ext_vector_type(8)))  __bf16 v8bf;
typedef __attribute__((ext_vector_type(8)))  float  v8f;
typedef int v4i_t __attribute__((vector_size(16)));   // GCC-style int4 (matches builtin proto V4i)

#define BDIM  16
#define NTOK  4096
#define CDIM  512
#define HDIM  8
#define DHEAD 64

// ---- CDNA5 async global->LDS copy (ASYNCcnt path), guarded ---------------
#if __has_builtin(__builtin_amdgcn_global_load_async_to_lds_b128)
#define USE_ASYNC_LDS 1
#else
#define USE_ASYNC_LDS 0
#endif

__device__ __forceinline__ void async_copy_b128(const void* g, void* l) {
#if USE_ASYNC_LDS
    __builtin_amdgcn_global_load_async_to_lds_b128(
        (__attribute__((address_space(1))) v4i_t*)g,
        (__attribute__((address_space(3))) v4i_t*)l, 0, 0);
#else
    *(v8bf*)l = *(const v8bf*)g;
#endif
}

__device__ __forceinline__ void async_wait0() {
#if USE_ASYNC_LDS
#if __has_builtin(__builtin_amdgcn_s_wait_asynccnt)
    __builtin_amdgcn_s_wait_asynccnt(0);
#else
    asm volatile("s_wait_asynccnt 0x0" ::: "memory");
#endif
#endif
}

union FragA { v16bf v; v8bf h[2]; };

__device__ __forceinline__ v8f v8f_zero() {
    v8f z;
    #pragma unroll
    for (int i = 0; i < 8; ++i) z[i] = 0.0f;
    return z;
}

__device__ __forceinline__ v8f wmma_bf16(const v16bf& a, const v16bf& b, const v8f& c) {
    // D = A(16x32 bf16) x B(32x16 bf16) + C(16x16 f32)
    return __builtin_amdgcn_wmma_f32_16x16x32_bf16(false, a, false, b, (short)0, c, false, false);
}

// ---------------------------------------------------------------------------
// Kernel 1: qkv = x @ qkv_w^T  (M=65536, N=1536, K=512)
// Output routed to: q,k in [B,H,64,N] bf16 (token-contiguous rows),
//                   v   in [B,N,C]    bf16 (token-major).
// 128x128 block tile, 8 waves each computing 2(M)x4(N) 16x16 subtiles.
// Single-barrier double-buffered pipeline.
// ---------------------------------------------------------------------------
__global__ __launch_bounds__(256)
void qkv_gemm_kernel(const float* __restrict__ x, const float* __restrict__ w,
                     bf16_t* __restrict__ qb, bf16_t* __restrict__ kb,
                     bf16_t* __restrict__ vtok)
{
    __shared__ __align__(16) bf16_t As[2][128][40];
    __shared__ __align__(16) bf16_t Bs[2][128][40];

    const int t    = threadIdx.x;
    const int lane = t & 31;
    const int wave = t >> 5;
    const int hi16 = (lane >> 4) & 1;
    const int ln16 = lane & 15;

    const int mbase = blockIdx.y * 128;   // token tile
    const int obase = blockIdx.x * 128;   // output-channel tile (0..1535)

    const int row  = t >> 1;              // 0..127
    const int half = (t & 1) * 16;        // 0 or 16 (k offset within 32-chunk)

    const float* xsrc = x + (size_t)(mbase + row) * CDIM + half;
    const float* wsrc = w + (size_t)(obase + row) * CDIM + half;

    v8f acc[2][4];
    #pragma unroll
    for (int i = 0; i < 2; ++i)
        #pragma unroll
        for (int j = 0; j < 4; ++j) acc[i][j] = v8f_zero();

    float4 ax[4], bx[4];
    #pragma unroll
    for (int q4 = 0; q4 < 4; ++q4) {
        ax[q4] = *(const float4*)(xsrc + q4 * 4);
        bx[q4] = *(const float4*)(wsrc + q4 * 4);
    }

    auto stage = [&](int bsel) {
        #pragma unroll
        for (int h2 = 0; h2 < 2; ++h2) {
            const float* fa = (const float*)(&ax[h2 * 2]);
            const float* fb = (const float*)(&bx[h2 * 2]);
            v8bf pa, pb;
            #pragma unroll
            for (int e = 0; e < 8; ++e) { pa[e] = (bf16_t)fa[e]; pb[e] = (bf16_t)fb[e]; }
            *(v8bf*)&As[bsel][row][half + h2 * 8] = pa;
            *(v8bf*)&Bs[bsel][row][half + h2 * 8] = pb;
        }
    };

    stage(0);
    __syncthreads();

    const int waveM = wave >> 1;      // 0..3 -> rows waveM*32 + {0,16}
    const int waveN = wave & 1;       // 0..1 -> cols waveN*64 + {0,16,32,48}
    const int mrow0 = waveM * 32;
    const int nrow0 = waveN * 64;

    int buf = 0;                      // invariant: LDS[buf] holds chunk kc
    for (int kc = 0; kc < 16; ++kc) {
        if (kc < 15) {
            const float* xa = xsrc + (kc + 1) * 32;
            const float* wb = wsrc + (kc + 1) * 32;
            #pragma unroll
            for (int q4 = 0; q4 < 4; ++q4) {
                ax[q4] = *(const float4*)(xa + q4 * 4);
                bx[q4] = *(const float4*)(wb + q4 * 4);
            }
        }
        if (kc < 14) {
            __builtin_prefetch(xsrc + (kc + 2) * 32, 0, 1);
            __builtin_prefetch(wsrc + (kc + 2) * 32, 0, 1);
        }

        FragA afr[2], bfr[4];
        #pragma unroll
        for (int ms = 0; ms < 2; ++ms) {
            const int m  = mrow0 + ms * 16 + ln16;
            const int ko = hi16 * 8;
            afr[ms].h[0] = *(const v8bf*)&As[buf][m][ko];
            afr[ms].h[1] = *(const v8bf*)&As[buf][m][ko + 16];
        }
        #pragma unroll
        for (int ns = 0; ns < 4; ++ns) {
            const int n  = nrow0 + ns * 16 + ln16;
            const int ko = hi16 * 8;
            bfr[ns].h[0] = *(const v8bf*)&Bs[buf][n][ko];
            bfr[ns].h[1] = *(const v8bf*)&Bs[buf][n][ko + 16];
        }
        #pragma unroll
        for (int ms = 0; ms < 2; ++ms)
            #pragma unroll
            for (int ns = 0; ns < 4; ++ns)
                acc[ms][ns] = wmma_bf16(afr[ms].v, bfr[ns].v, acc[ms][ns]);

        if (kc < 15) {
            stage(buf ^ 1);           // writes buf^1; reads above were from buf
            __syncthreads();
            buf ^= 1;
        }
    }

    // Epilogue: route to q / k ([B,H,64,N]) or v ([B,N,C])
    #pragma unroll
    for (int ms = 0; ms < 2; ++ms) {
        const int tok0 = mbase + mrow0 + ms * 16 + hi16 * 8;  // + r (consecutive)
        const int bidx = tok0 >> 12;
        const int n0   = tok0 & 4095;
        #pragma unroll
        for (int ns = 0; ns < 4; ++ns) {
            const int o    = obase + nrow0 + ns * 16 + ln16;
            const int qkvi = o >> 9;          // 0=q,1=k,2=v (uniform per subtile)
            const int hh   = (o >> 6) & 7;
            const int dch  = o & 63;
            if (qkvi < 2) {
                bf16_t* dst = (qkvi == 0 ? qb : kb)
                            + ((size_t)(bidx * HDIM + hh) * DHEAD + dch) * NTOK + n0;
                v8bf p;
                #pragma unroll
                for (int r = 0; r < 8; ++r) p[r] = (bf16_t)acc[ms][ns][r];
                *(v8bf*)dst = p;
            } else {
                const int c = o - 1024;       // = hh*64 + dch
                #pragma unroll
                for (int r = 0; r < 8; ++r)
                    vtok[((size_t)bidx * NTOK + (n0 + r)) * CDIM + c] = (bf16_t)acc[ms][ns][r];
            }
        }
    }
}

// ---------------------------------------------------------------------------
// Kernel 2: rscale[row] = 1 / max(||row||_2, eps)  over N=4096
// rows 0..8191 = q (bh*64+d), rows 8192..16383 = k
// ---------------------------------------------------------------------------
__global__ __launch_bounds__(128)
void rownorm_kernel(const bf16_t* __restrict__ q, const bf16_t* __restrict__ k,
                    float* __restrict__ rscale)
{
    __shared__ float red[128];
    const int rowid = blockIdx.x;
    const bf16_t* p = (rowid < 8192 ? q : k) + (size_t)(rowid & 8191) * NTOK;

    float s = 0.0f;
    for (int i = threadIdx.x * 8; i < NTOK; i += 128 * 8) {
        v8bf v = *(const v8bf*)(p + i);
        #pragma unroll
        for (int e = 0; e < 8; ++e) { float f = (float)v[e]; s += f * f; }
    }
    red[threadIdx.x] = s;
    __syncthreads();
    #pragma unroll
    for (int off = 64; off > 0; off >>= 1) {
        if (threadIdx.x < off) red[threadIdx.x] += red[threadIdx.x + off];
        __syncthreads();
    }
    if (threadIdx.x == 0)
        rscale[rowid] = 1.0f / fmaxf(sqrtf(red[0]), 1e-12f);
}

// ---------------------------------------------------------------------------
// Kernel 3: per (b,h):  S = q k^T (64x64, K=4096) -> scale/softmax -> attn v
// q,k in [B,H,64,N]; v,ctx token-major [B,N,C].
// ---------------------------------------------------------------------------
__global__ __launch_bounds__(256)
void attn_kernel(const bf16_t* __restrict__ q, const bf16_t* __restrict__ k,
                 const bf16_t* __restrict__ vtok, const float* __restrict__ rscale,
                 const float* __restrict__ temperature, bf16_t* __restrict__ ctx)
{
    __shared__ float  S[64][65];
    __shared__ __align__(16) bf16_t Ab[64][72];
    __shared__ float  rks[64];

    const int bh   = blockIdx.x;          // b*H + h
    const int b    = bh >> 3;
    const int h    = bh & 7;
    const int t    = threadIdx.x;
    const int lane = t & 31;
    const int wave = t >> 5;
    const int hi16 = (lane >> 4) & 1;
    const int ln16 = lane & 15;

    const bf16_t* qrow = q + (size_t)bh * DHEAD * NTOK;
    const bf16_t* krow = k + (size_t)bh * DHEAD * NTOK;

    // ---- Phase A: covariance S = q @ k^T, 16 subtiles over 8 waves -------
    const int mt = wave >> 2;             // 0..1 -> mtiles {mt, mt+2}
    const int nt = wave & 3;              // 0..3
    v8f accS[2]; accS[0] = v8f_zero(); accS[1] = v8f_zero();

    const bf16_t* aptr0 = qrow + (size_t)((mt    ) * 16 + ln16) * NTOK + hi16 * 8;
    const bf16_t* aptr1 = qrow + (size_t)((mt + 2) * 16 + ln16) * NTOK + hi16 * 8;
    const bf16_t* bptr  = krow + (size_t)(nt * 16 + ln16) * NTOK + hi16 * 8;

    for (int kk = 0; kk < NTOK; kk += 32) {
        FragA a0, a1, bb;
        a0.h[0] = *(const v8bf*)(aptr0 + kk);
        a0.h[1] = *(const v8bf*)(aptr0 + kk + 16);
        a1.h[0] = *(const v8bf*)(aptr1 + kk);
        a1.h[1] = *(const v8bf*)(aptr1 + kk + 16);
        bb.h[0] = *(const v8bf*)(bptr + kk);
        bb.h[1] = *(const v8bf*)(bptr + kk + 16);
        accS[0] = wmma_bf16(a0.v, bb.v, accS[0]);
        accS[1] = wmma_bf16(a1.v, bb.v, accS[1]);
        if (kk + 64 < NTOK) __builtin_prefetch(bptr + kk + 64, 0, 1);
    }
    #pragma unroll
    for (int mi = 0; mi < 2; ++mi) {
        const int mrow = (mt + mi * 2) * 16 + hi16 * 8;
        #pragma unroll
        for (int r = 0; r < 8; ++r) S[mrow + r][nt * 16 + ln16] = accS[mi][r];
    }
    if (t < 64) rks[t] = rscale[8192 + bh * 64 + t];
    __syncthreads();

    // ---- Phase B: scale by rq*rk*temperature, softmax over rows ----------
    if (t < 64) {
        const float rq = rscale[bh * 64 + t] * temperature[h];
        float mx = -3.0e38f;
        for (int j = 0; j < 64; ++j) {
            float v = S[t][j] * rq * rks[j];
            S[t][j] = v;
            mx = fmaxf(mx, v);
        }
        float sum = 0.0f;
        for (int j = 0; j < 64; ++j) {
            float e = __expf(S[t][j] - mx);
            S[t][j] = e;
            sum += e;
        }
        const float inv = 1.0f / sum;
        for (int j = 0; j < 64; ++j) Ab[t][j] = (bf16_t)(S[t][j] * inv);
    }
    __syncthreads();

    // ---- Phase C: out = attn @ v  (M=64, K=64, N=4096) -------------------
    FragA afr[4][2];
    #pragma unroll
    for (int mi = 0; mi < 4; ++mi)
        #pragma unroll
        for (int kc = 0; kc < 2; ++kc) {
            afr[mi][kc].h[0] = *(const v8bf*)&Ab[mi * 16 + ln16][kc * 32 + hi16 * 8];
            afr[mi][kc].h[1] = *(const v8bf*)&Ab[mi * 16 + ln16][kc * 32 + hi16 * 8 + 16];
        }

    const bf16_t* vbase = vtok + (size_t)b * NTOK * CDIM + h * 64;
    bf16_t*       cbase = ctx  + (size_t)b * NTOK * CDIM + h * 64;

    for (int ntile = wave; ntile < 256; ntile += 8) {
        const int token = ntile * 16 + ln16;
        const bf16_t* vp = vbase + (size_t)token * CDIM;
        FragA bfr[2];
        #pragma unroll
        for (int kc = 0; kc < 2; ++kc) {
            bfr[kc].h[0] = *(const v8bf*)(vp + kc * 32 + hi16 * 8);
            bfr[kc].h[1] = *(const v8bf*)(vp + kc * 32 + hi16 * 8 + 16);
        }
        v8f acc[4];
        #pragma unroll
        for (int mi = 0; mi < 4; ++mi) {
            acc[mi] = v8f_zero();
            #pragma unroll
            for (int kc = 0; kc < 2; ++kc)
                acc[mi] = wmma_bf16(afr[mi][kc].v, bfr[kc].v, acc[mi]);
        }
        #pragma unroll
        for (int mi = 0; mi < 4; ++mi) {
            v8bf p;
            #pragma unroll
            for (int r = 0; r < 8; ++r) p[r] = (bf16_t)acc[mi][r];
            *(v8bf*)(cbase + (size_t)token * CDIM + mi * 16 + hi16 * 8) = p;
        }
    }
}

// ---------------------------------------------------------------------------
// Kernel 4: out = ctx @ proj_w^T + proj_b  (M=65536, N=512, K=512), f32 out
// A (ctx, bf16) staged via async global->LDS (ASYNCcnt); B (f32 w) via regs+cvt.
// ---------------------------------------------------------------------------
__global__ __launch_bounds__(256)
void proj_gemm_kernel(const bf16_t* __restrict__ ctx, const float* __restrict__ w,
                      const float* __restrict__ bias, float* __restrict__ out)
{
    __shared__ __align__(16) bf16_t As[2][128][40];
    __shared__ __align__(16) bf16_t Bs[2][128][40];

    const int t    = threadIdx.x;
    const int lane = t & 31;
    const int wave = t >> 5;
    const int hi16 = (lane >> 4) & 1;
    const int ln16 = lane & 15;

    const int mbase = blockIdx.y * 128;
    const int obase = blockIdx.x * 128;
    const int row   = t >> 1;
    const int half  = (t & 1) * 16;

    const bf16_t* asrc = ctx + (size_t)(mbase + row) * CDIM + half;
    const float*  wsrc = w   + (size_t)(obase + row) * CDIM + half;

    v8f acc[2][4];
    #pragma unroll
    for (int i = 0; i < 2; ++i)
        #pragma unroll
        for (int j = 0; j < 4; ++j) acc[i][j] = v8f_zero();

    float4 bx[4];

    auto asyncA = [&](int bsel, int kc) {
        const bf16_t* g = asrc + kc * 32;
        async_copy_b128(g,     &As[bsel][row][half]);
        async_copy_b128(g + 8, &As[bsel][row][half + 8]);
    };
    auto loadB = [&](int kc) {
        const float* wb = wsrc + kc * 32;
        #pragma unroll
        for (int q4 = 0; q4 < 4; ++q4) bx[q4] = *(const float4*)(wb + q4 * 4);
    };
    auto stageB = [&](int bsel) {
        #pragma unroll
        for (int h2 = 0; h2 < 2; ++h2) {
            const float* fb = (const float*)(&bx[h2 * 2]);
            v8bf pb;
            #pragma unroll
            for (int e = 0; e < 8; ++e) pb[e] = (bf16_t)fb[e];
            *(v8bf*)&Bs[bsel][row][half + h2 * 8] = pb;
        }
    };

    // prologue: chunk 0 into buffer 0
    asyncA(0, 0);
    loadB(0);
    stageB(0);
    async_wait0();
    __syncthreads();

    const int waveM = wave >> 1;
    const int waveN = wave & 1;
    const int mrow0 = waveM * 32;
    const int nrow0 = waveN * 64;

    int buf = 0;                      // invariant: LDS[buf] holds chunk kc
    for (int kc = 0; kc < 16; ++kc) {
        if (kc < 15) {
            asyncA(buf ^ 1, kc + 1);  // async writes to the other buffer
            loadB(kc + 1);
        }
        if (kc < 14) {
            __builtin_prefetch(asrc + (kc + 2) * 32, 0, 1);
            __builtin_prefetch(wsrc + (kc + 2) * 32, 0, 1);
        }

        FragA afr[2], bfr[4];
        #pragma unroll
        for (int ms = 0; ms < 2; ++ms) {
            const int m  = mrow0 + ms * 16 + ln16;
            const int ko = hi16 * 8;
            afr[ms].h[0] = *(const v8bf*)&As[buf][m][ko];
            afr[ms].h[1] = *(const v8bf*)&As[buf][m][ko + 16];
        }
        #pragma unroll
        for (int ns = 0; ns < 4; ++ns) {
            const int n  = nrow0 + ns * 16 + ln16;
            const int ko = hi16 * 8;
            bfr[ns].h[0] = *(const v8bf*)&Bs[buf][n][ko];
            bfr[ns].h[1] = *(const v8bf*)&Bs[buf][n][ko + 16];
        }
        #pragma unroll
        for (int ms = 0; ms < 2; ++ms)
            #pragma unroll
            for (int ns = 0; ns < 4; ++ns)
                acc[ms][ns] = wmma_bf16(afr[ms].v, bfr[ns].v, acc[ms][ns]);

        if (kc < 15) {
            stageB(buf ^ 1);
            async_wait0();            // A-tile async writes to buf^1 landed
            __syncthreads();
            buf ^= 1;
        }
    }

    #pragma unroll
    for (int ns = 0; ns < 4; ++ns) {
        const int c = obase + nrow0 + ns * 16 + ln16;
        const float bc = bias[c];
        #pragma unroll
        for (int ms = 0; ms < 2; ++ms) {
            const int tok0 = mbase + mrow0 + ms * 16 + hi16 * 8;
            #pragma unroll
            for (int r = 0; r < 8; ++r)
                out[(size_t)(tok0 + r) * CDIM + c] = acc[ms][ns][r] + bc;
        }
    }
}

// ---------------------------------------------------------------------------
extern "C" void kernel_launch(void* const* d_in, const int* in_sizes, int n_in,
                              void* d_out, int out_size, void* d_ws, size_t ws_size,
                              hipStream_t stream) {
    (void)in_sizes; (void)n_in; (void)out_size; (void)ws_size;
    const float* x           = (const float*)d_in[0];
    const float* qkv_w       = (const float*)d_in[1];
    const float* temperature = (const float*)d_in[2];
    const float* proj_w      = (const float*)d_in[3];
    const float* proj_b      = (const float*)d_in[4];
    float* out = (float*)d_out;

    // Workspace layout (bf16 intermediates): 4 x 64MB + scales
    char* ws = (char*)d_ws;
    const size_t SEG = (size_t)BDIM * HDIM * DHEAD * NTOK * sizeof(bf16_t); // 64 MiB
    bf16_t* qb     = (bf16_t*)(ws);
    bf16_t* kb     = (bf16_t*)(ws + SEG);
    bf16_t* vtok   = (bf16_t*)(ws + 2 * SEG);
    bf16_t* ctxb   = (bf16_t*)(ws + 3 * SEG);
    float*  rscale = (float*)(ws + 4 * SEG);

    qkv_gemm_kernel<<<dim3(12, 512), 256, 0, stream>>>(x, qkv_w, qb, kb, vtok);
    rownorm_kernel<<<dim3(16384), 128, 0, stream>>>(qb, kb, rscale);
    attn_kernel<<<dim3(BDIM * HDIM), 256, 0, stream>>>(qb, kb, vtok, rscale, temperature, ctxb);
    proj_gemm_kernel<<<dim3(4, 512), 256, 0, stream>>>(ctxb, proj_w, proj_b, out);
}
